// _Memory_Block_44126493999738
// MI455X (gfx1250) — matrix operand order repaired and checked by reading, not compiled, not run on hardware
//
#include <hip/hip_runtime.h>

// ---------------- problem constants ----------------
#define NPIX 65536            // 16*64*64 pixels
#define CCH  256              // channels
#define KCB  512              // codebook size
#define HW   4096             // 64*64
#define RATE 0.999f

typedef __attribute__((ext_vector_type(16))) _Float16 v16h;
typedef __attribute__((ext_vector_type(8)))  _Float16 v8h;
typedef __attribute__((ext_vector_type(8)))  float    v8f;

#if __has_builtin(__builtin_amdgcn_global_load_async_to_lds_b128)
#define HAVE_ASYNC_LDS 1
typedef int i128 __attribute__((vector_size(16)));      // b128 payload type per builtin decl
typedef __attribute__((address_space(1))) i128* g128p;  // global src
typedef __attribute__((address_space(3))) i128* l128p;  // LDS dst
#endif

static __device__ __forceinline__ float wsum(float v) {
#pragma unroll
  for (int m = 16; m >= 1; m >>= 1) v += __shfl_xor(v, m, 32);
  return v;
}
static __device__ __forceinline__ float wmax(float v) {
#pragma unroll
  for (int m = 16; m >= 1; m >>= 1) v = fmaxf(v, __shfl_xor(v, m, 32));
  return v;
}

// Stage one contiguous 16x256 f16 A-tile (8 KB) from global into LDS.
// Async path: 16 x (32 lanes x B128) coalesced transfers tracked by ASYNCcnt.
static __device__ __forceinline__ void stage_A(const _Float16* __restrict__ g,
                                               _Float16* ls, int lane) {
#ifdef HAVE_ASYNC_LDS
#pragma unroll
  for (int i = 0; i < 16; ++i) {
    __builtin_amdgcn_global_load_async_to_lds_b128(
        (g128p)(g + i * 256 + lane * 8),
        (l128p)(ls + i * 256 + lane * 8), 0, 0);
  }
#if __has_builtin(__builtin_amdgcn_s_wait_asynccnt)
  __builtin_amdgcn_s_wait_asynccnt(0);
#else
  asm volatile("s_wait_asynccnt 0" ::: "memory");
#endif
  asm volatile("s_wait_dscnt 0" ::: "memory");
#else
#pragma unroll
  for (int i = 0; i < 16; ++i) {
    v8h t = *(const v8h*)(g + i * 256 + lane * 8);
    *(v8h*)(ls + i * 256 + lane * 8) = t;
  }
  asm volatile("s_wait_dscnt 0" ::: "memory");
#endif
}

// Build WMMA A-fragment (16x32 f16) from an LDS-resident row-major 16xCCH tile.
// ISA layout: lanes 0-15 hold M=lane, K = kb+0..7 (v0..v3) and kb+16..23 (v4..v7);
// lanes 16-31 hold M=lane-16, K = kb+8..15 and kb+24..31.
static __device__ __forceinline__ v16h afrag_lds(const _Float16* ls, int kb, int m, int hi) {
  const _Float16* rp = ls + m * CCH + kb + hi * 8;
  v8h lo = *(const v8h*)rp;
  v8h h2 = *(const v8h*)(rp + 16);
  v16h a;
#pragma unroll
  for (int i = 0; i < 8; ++i) { a[i] = lo[i]; a[i + 8] = h2[i]; }
  return a;
}

// ---------------- pass 1: normalize x -> xn (f16, n x 256) ----------------
__global__ void k_norm_x(const float* __restrict__ x, _Float16* __restrict__ xn) {
  int wid = threadIdx.x >> 5, lane = threadIdx.x & 31;
  int n = blockIdx.x * 8 + wid;
  size_t img = (size_t)(n >> 12) * ((size_t)CCH * HW);
  int hw = n & (HW - 1);
  const float* px = x + img + hw;
  int c0 = lane * 8;
  float v[8]; float s = 0.f;
#pragma unroll
  for (int j = 0; j < 8; ++j) { v[j] = px[(size_t)(c0 + j) * HW]; s += v[j] * v[j]; }
  s = wsum(s);
  float inv = 1.0f / fmaxf(sqrtf(s), 1e-12f);
  v8h st;
#pragma unroll
  for (int j = 0; j < 8; ++j) st[j] = (_Float16)(v[j] * inv);
  *(v8h*)(xn + (size_t)n * CCH + c0) = st;
}

// ---------------- pass 1b: normalize m -> mnT (f16, 256 x 512) ----------------
__global__ void k_norm_m(const float* __restrict__ m, _Float16* __restrict__ mnT) {
  int wid = threadIdx.x >> 5, lane = threadIdx.x & 31;
  int k = blockIdx.x * 8 + wid;
  int c0 = lane * 8;
  const float* row = m + (size_t)k * CCH + c0;
  float v[8]; float s = 0.f;
#pragma unroll
  for (int j = 0; j < 8; ++j) { v[j] = row[j]; s += v[j] * v[j]; }
  s = wsum(s);
  float inv = 1.0f / fmaxf(sqrtf(s), 1e-12f);
#pragma unroll
  for (int j = 0; j < 8; ++j) mnT[(size_t)(c0 + j) * KCB + k] = (_Float16)(v[j] * inv);
}

__global__ void k_zero(float* __restrict__ p, int n) {
  int i = blockIdx.x * 256 + threadIdx.x;
  if (i < n) p[i] = 0.f;
}

// ---------------- pass 2: score1 = xn @ mnT, per-row argmax ----------------
__global__ void k_argmax(const _Float16* __restrict__ xn, const _Float16* __restrict__ mnT,
                         int* __restrict__ ind) {
  __shared__ _Float16 Ast[8 * 16 * 256];       // 64 KB: one 8 KB A-tile per wave
  int wid = threadIdx.x >> 5, lane = threadIdx.x & 31;
  int tile = blockIdx.x * 8 + wid;             // 4096 tiles of 16 pixels
  int nb = tile * 16;
  int m = lane & 15, hi = lane >> 4;
  _Float16* as = Ast + wid * (16 * 256);
  stage_A(xn + (size_t)nb * CCH, as, lane);

  v16h A[8];
#pragma unroll
  for (int kk = 0; kk < 8; ++kk) A[kk] = afrag_lds(as, kk * 32, m, hi);
  float best[8]; int bidx[8];
#pragma unroll
  for (int v = 0; v < 8; ++v) { best[v] = -1e30f; bidx[v] = 0; }
  for (int ct = 0; ct < 32; ++ct) {
    __builtin_prefetch(mnT + (size_t)lane * KCB + (ct + 1) * 16, 0, 1);
    v8f acc = {};
#pragma unroll
    for (int kk = 0; kk < 8; ++kk) {
      // B-frag: lane L holds col (L%16), K = kk*32 + hi*8 + 0..7 and + 16 + hi*8 + 0..7
      v16h b;
#pragma unroll
      for (int i = 0; i < 8; ++i) { b[i] = mnT[(size_t)(kk * 32 + hi * 8 + i) * KCB + ct * 16 + m]; b[i + 8] = mnT[(size_t)(kk * 32 + 16 + hi * 8 + i) * KCB + ct * 16 + m]; }
      acc = __builtin_amdgcn_wmma_f32_16x16x32_f16(false, A[kk], false, b,
                                                   (short)0, acc, false, false);
    }
    int col = ct * 16 + m;
#pragma unroll
    for (int v = 0; v < 8; ++v)
      if (acc[v] > best[v]) { best[v] = acc[v]; bidx[v] = col; }
  }
#pragma unroll
  for (int v = 0; v < 8; ++v) {
    float bv = best[v]; int bi = bidx[v];
#pragma unroll
    for (int off = 8; off >= 1; off >>= 1) {
      float ov = __shfl_xor(bv, off, 32);
      int   oi = __shfl_xor(bi, off, 32);
      if (ov > bv) { bv = ov; bi = oi; }
    }
    if (m == 0) ind[nb + hi * 8 + v] = bi;     // rows v (hi=0) / v+8 (hi=1)
  }
}

// ---------------- pass 3: segment sums of raw x ----------------
__global__ void k_segsum(const float* __restrict__ x, const int* __restrict__ ind,
                         float* __restrict__ esum, float* __restrict__ cnt) {
  int wid = threadIdx.x >> 5, lane = threadIdx.x & 31;
  int n = blockIdx.x * 8 + wid;
  int k = ind[n];
  size_t img = (size_t)(n >> 12) * ((size_t)CCH * HW);
  int hw = n & (HW - 1);
  const float* px = x + img + hw;
  int c0 = lane * 8;
#pragma unroll
  for (int j = 0; j < 8; ++j)
    atomicAdd(&esum[(size_t)k * CCH + c0 + j], px[(size_t)(c0 + j) * HW]);
  if (lane == 0) atomicAdd(&cnt[k], 1.0f);
}

// ---------------- pass 4: EMA update, new_m (f16 KxC) + mn2T (f16 CxK) ----------------
__global__ void k_update(const float* __restrict__ m, const float* __restrict__ esum,
                         const float* __restrict__ cnt, _Float16* __restrict__ nm,
                         _Float16* __restrict__ mn2T) {
  int wid = threadIdx.x >> 5, lane = threadIdx.x & 31;
  int k = blockIdx.x * 8 + wid;
  int c0 = lane * 8;
  float cc = cnt[k] + 1e-6f;
  float v[8]; float s = 0.f;
#pragma unroll
  for (int j = 0; j < 8; ++j) {
    size_t i = (size_t)k * CCH + c0 + j;
    v[j] = m[i] * RATE + (esum[i] / cc) * (1.0f - RATE);
    s += v[j] * v[j];
  }
  s = wsum(s);
  float inv = 1.0f / fmaxf(sqrtf(s), 1e-12f);
  v8h st;
#pragma unroll
  for (int j = 0; j < 8; ++j) st[j] = (_Float16)v[j];
  *(v8h*)(nm + (size_t)k * CCH + c0) = st;
#pragma unroll
  for (int j = 0; j < 8; ++j) mn2T[(size_t)(c0 + j) * KCB + k] = (_Float16)(v[j] * inv);
}

// ---------------- pass 5: score2 -> softmax -> out, fused ----------------
__global__ void __launch_bounds__(64)
k_score2_out(const _Float16* __restrict__ xn, const _Float16* __restrict__ mn2T,
             const _Float16* __restrict__ nm, float* __restrict__ out1,
             float* __restrict__ out2) {
  __shared__ float Sall[2 * 16 * 512];                 // 64 KB, one 16x512 slab per wave
  int wid = threadIdx.x >> 5, lane = threadIdx.x & 31;
  float* S = Sall + wid * (16 * 512);
  int tile = blockIdx.x * 2 + wid;
  int nb = tile * 16;
  int m = lane & 15, hi = lane >> 4;

  // Stage A-tile into (soon to be overwritten) score slab, lift to registers.
  stage_A(xn + (size_t)nb * CCH, (_Float16*)S, lane);
  v16h A[8];
#pragma unroll
  for (int kk = 0; kk < 8; ++kk) A[kk] = afrag_lds((const _Float16*)S, kk * 32, m, hi);

  // GEMM2: score2 tile 16x512 into LDS
  for (int ct = 0; ct < 32; ++ct) {
    __builtin_prefetch(mn2T + (size_t)lane * KCB + (ct + 1) * 16, 0, 1);
    v8f acc = {};
#pragma unroll
    for (int kk = 0; kk < 8; ++kk) {
      v16h b;
#pragma unroll
      for (int i = 0; i < 8; ++i) { b[i] = mn2T[(size_t)(kk * 32 + hi * 8 + i) * KCB + ct * 16 + m]; b[i + 8] = mn2T[(size_t)(kk * 32 + 16 + hi * 8 + i) * KCB + ct * 16 + m]; }
      acc = __builtin_amdgcn_wmma_f32_16x16x32_f16(false, A[kk], false, b,
                                                   (short)0, acc, false, false);
    }
    int col = ct * 16 + m;
#pragma unroll
    for (int v = 0; v < 8; ++v) S[(hi * 8 + v) * 512 + col] = acc[v];
  }
  asm volatile("s_wait_dscnt 0" ::: "memory");

  // write raw score2 (coalesced from LDS)
  for (int r = 0; r < 16; ++r) {
    const float* src = S + r * 512;
    float* dst = out2 + (size_t)(nb + r) * KCB;
    for (int j = lane; j < 512; j += 32) dst[j] = src[j];
  }

  // softmax per row, in place
  for (int r = 0; r < 16; ++r) {
    float* row = S + r * 512;
    float mx = -1e30f;
    for (int j = lane; j < 512; j += 32) mx = fmaxf(mx, row[j]);
    mx = wmax(mx);
    float sum = 0.f;
    for (int j = lane; j < 512; j += 32) { float e = __expf(row[j] - mx); row[j] = e; sum += e; }
    sum = wsum(sum);
    float inv = 1.0f / sum;
    for (int j = lane; j < 512; j += 32) row[j] *= inv;
  }
  asm volatile("s_wait_dscnt 0" ::: "memory");

  // GEMM3: out tile (16x256) = P(16x512) @ new_m(512x256)
  for (int ct2 = 0; ct2 < 16; ++ct2) {
    v8f acc = {};
#pragma unroll
    for (int q = 0; q < 16; ++q) {
      const float* rp = S + m * 512 + q * 32 + hi * 8;   // A-frag from probs
      v16h a;
#pragma unroll
      for (int i = 0; i < 8; ++i) { a[i] = (_Float16)rp[i]; a[i + 8] = (_Float16)rp[16 + i]; }
      v16h b;
#pragma unroll
      for (int i = 0; i < 8; ++i) { b[i] = nm[(size_t)(q * 32 + hi * 8 + i) * CCH + ct2 * 16 + m]; b[i + 8] = nm[(size_t)(q * 32 + 16 + hi * 8 + i) * CCH + ct2 * 16 + m]; }
      acc = __builtin_amdgcn_wmma_f32_16x16x32_f16(false, a, false, b,
                                                   (short)0, acc, false, false);
    }
    int c = ct2 * 16 + m;
#pragma unroll
    for (int v = 0; v < 8; ++v) {
      int n = nb + hi * 8 + v;
      out1[(size_t)(n >> 12) * ((size_t)CCH * HW) + (size_t)c * HW + (n & (HW - 1))] = acc[v];
    }
  }
}

// ---------------- host launcher ----------------
extern "C" void kernel_launch(void* const* d_in, const int* in_sizes, int n_in,
                              void* d_out, int out_size, void* d_ws, size_t ws_size,
                              hipStream_t stream) {
  const float* x = (const float*)d_in[0];   // (16,256,64,64)
  const float* m = (const float*)d_in[1];   // (512,256)
  float* out1 = (float*)d_out;                              // (16,256,64,64)
  float* out2 = out1 + (size_t)NPIX * CCH;                  // (65536,512)

  char* ws = (char*)d_ws;
  _Float16* xn   = (_Float16*)(ws);                         // 33,554,432 B
  _Float16* mnT  = (_Float16*)(ws + 33554432);              //    262,144 B
  _Float16* mn2T = (_Float16*)(ws + 33816576);              //    262,144 B
  _Float16* nm   = (_Float16*)(ws + 34078720);              //    262,144 B
  float*    esum = (float*)   (ws + 34340864);              //    524,288 B
  float*    cnt  = (float*)   (ws + 34865152);              //      2,048 B
  int*      ind  = (int*)     (ws + 34867200);              //    262,144 B

  k_norm_x<<<NPIX / 8, 256, 0, stream>>>(x, xn);
  k_norm_m<<<KCB / 8, 256, 0, stream>>>(m, mnT);
  k_zero<<<(131584 + 255) / 256, 256, 0, stream>>>(esum, 131584);  // esum+cnt adjacent
  k_argmax<<<(NPIX / 16) / 8, 256, 0, stream>>>(xn, mnT, ind);
  k_segsum<<<NPIX / 8, 256, 0, stream>>>(x, ind, esum, cnt);
  k_update<<<KCB / 8, 256, 0, stream>>>(m, esum, cnt, nm, mn2T);
  k_score2_out<<<(NPIX / 16) / 2, 64, 0, stream>>>(xn, mn2T, nm, out1, out2);
}
